// GroupedQueryAttention_43413529428365
// MI455X (gfx1250) — matrix operand order repaired and checked by reading, not compiled, not run on hardware
//
#include <hip/hip_runtime.h>
#include <hip/hip_bf16.h>
#include <math.h>

typedef __attribute__((ext_vector_type(16))) __bf16 v16bf;
typedef __attribute__((ext_vector_type(8)))  __bf16 v8bf;
typedef __attribute__((ext_vector_type(8)))  float  v8f;

#define S_LEN 2048
#define EMB   2048
#define KVD   512
#define HD    64
#define NH    32
#define NKV   8

union AFrag { v16bf v; v8bf h[2]; };

// ---------------- fp32 -> bf16 cast ----------------
__global__ void cast_f32_bf16(const float* __restrict__ in, __bf16* __restrict__ out, int n) {
  int i = blockIdx.x * blockDim.x + threadIdx.x;
  if (i < n) out[i] = (__bf16)in[i];
}

// ---------------- V -> Vt [b, kv, d, s] ----------------
__global__ void transpose_v(const __bf16* __restrict__ v, __bf16* __restrict__ vt) {
  int i = blockIdx.x * blockDim.x + threadIdx.x;     // over 2*2048*512
  if (i >= 2 * S_LEN * KVD) return;
  int b   = i >> 20;
  int rem = i & ((1 << 20) - 1);
  int s   = rem >> 9;
  int c   = rem & 511;
  int kv  = c >> 6;
  int d   = c & 63;
  vt[(((size_t)b * NKV + kv) * HD + d) * S_LEN + s] = v[i];
}

// ---------------- bf16 GEMM: out = (A @ W^T + bias) * scale ----------------
// A: [M,K] bf16 row-major; W: [N,K] bf16 row-major (torch Linear layout).
// 128 threads = 4 waves; workgroup computes a 64x64 tile. The shared B tile
// (64 n-rows x 64 k) is staged in LDS via async global->LDS DMA, double
// buffered, so the next tile's copy overlaps the current tile's WMMAs.
// LDS rows padded to 72 elems (144B): fragment reads hit 16 distinct 4-bank
// groups -> conflict-free.
template<bool OUT_BF16>
__global__ __launch_bounds__(128, 1)
void gemm_bf16(const __bf16* __restrict__ A, const __bf16* __restrict__ W,
               const float* __restrict__ bias, void* __restrict__ out,
               int M, int N, int K, float scale) {
  __shared__ __bf16 ldsB[2][64][72];

  const int tid  = threadIdx.x;
  const int wave = tid >> 5;
  const int lane = tid & 31;
  const int lm = lane & 15;
  const int lg = lane >> 4;
  const int m0 = blockIdx.y * 64 + wave * 16;
  const int n0 = blockIdx.x * 64;

  const unsigned ldsBase = (unsigned)(size_t)(&ldsB[0][0][0]);

  auto copy_tile = [&](int k0, int buf) {
#pragma unroll
    for (int j = 0; j < 4; ++j) {
      int cch = j * 128 + tid;          // 512 chunks of 16B
      int n   = cch >> 3;
      int ke  = (cch & 7) * 8;          // element offset within 64-k row
      unsigned loff = ldsBase + (unsigned)(buf * (64 * 72 * 2) + n * 144 + ke * 2);
      const __bf16* gp = W + (size_t)(n0 + n) * K + k0 + ke;
      asm volatile("global_load_async_to_lds_b128 %0, %1, off"
                   :: "v"(loff), "v"(gp) : "memory");
    }
  };

  v8f c[4] = {v8f{}, v8f{}, v8f{}, v8f{}};
  const __bf16* arow = A + (size_t)(m0 + lm) * K;
  const int nsteps = K >> 6;

  copy_tile(0, 0);
  for (int i = 0; i < nsteps; ++i) {
    asm volatile("s_wait_asynccnt 0x0" ::: "memory");
    __syncthreads();                               // buf (i&1) ready on all waves
    if (i + 1 < nsteps) copy_tile((i + 1) << 6, (i + 1) & 1);

    const int k0 = i << 6;
    AFrag a[2];
#pragma unroll
    for (int cc = 0; cc < 2; ++cc) {
      a[cc].h[0] = *(const v8bf*)(arow + k0 + cc * 32 + lg * 16);
      a[cc].h[1] = *(const v8bf*)(arow + k0 + cc * 32 + lg * 16 + 8);
    }
#pragma unroll
    for (int t = 0; t < 4; ++t) {
#pragma unroll
      for (int cc = 0; cc < 2; ++cc) {
        v16bf bfrag = *(const v16bf*)(&ldsB[i & 1][t * 16 + lm][cc * 32 + lg * 16]);
        c[t] = __builtin_amdgcn_wmma_f32_16x16x32_bf16(
            false, a[cc].v, false, bfrag, (short)0, c[t], false, false);
      }
    }
  }

#pragma unroll
  for (int t = 0; t < 4; ++t) {
    int n = n0 + t * 16 + lm;
    float bn = bias[n];
#pragma unroll
    for (int r = 0; r < 8; ++r) {
      int row = m0 + r + 8 * lg;
      float val = (c[t][r] + bn) * scale;
      if (OUT_BF16) ((__bf16*)out)[(size_t)row * N + n] = (__bf16)val;
      else          ((float*) out)[(size_t)row * N + n] = val;
    }
  }
}

// ---------------- flash attention (causal, GQA) ----------------
// One wave32 per (batch, head, 16-query-row tile). Online softmax over key
// tiles of 32; QK^T and PV both via v_wmma_f32_16x16x32_bf16.
__global__ __launch_bounds__(32, 1)
void flash_gqa(const __bf16* __restrict__ q, const __bf16* __restrict__ k,
               const __bf16* __restrict__ vt, __bf16* __restrict__ o) {
  __shared__ __bf16 ldsP[16 * 32];   // P tile staging (C-layout -> A-layout)

  const int lane = threadIdx.x;
  const int lm = lane & 15;
  const int lg = lane >> 4;

  const int idx = blockIdx.x;            // 2 * 32 * 128 blocks
  const int mt = idx & 127;
  const int h  = (idx >> 7) & 31;
  const int b  = idx >> 12;
  const int kv = h >> 2;
  const int m0 = mt * 16;

  // Q fragments for head-dim chunks [0..31], [32..63] (scale already folded in)
  AFrag qa[2];
  const __bf16* qrow = q + (size_t)(b * S_LEN + m0 + lm) * EMB + h * HD;
#pragma unroll
  for (int cch = 0; cch < 2; ++cch) {
    qa[cch].h[0] = *(const v8bf*)(qrow + cch * 32 + lg * 16);
    qa[cch].h[1] = *(const v8bf*)(qrow + cch * 32 + lg * 16 + 8);
  }

  const __bf16* kbase  = k  + (size_t)b * S_LEN * KVD + kv * HD;
  const __bf16* vtbase = vt + ((size_t)(b * NKV + kv)) * HD * S_LEN;

  v8f acc[4] = {v8f{}, v8f{}, v8f{}, v8f{}};
  float rmax[8], rsum[8];
#pragma unroll
  for (int r = 0; r < 8; ++r) { rmax[r] = -__builtin_inff(); rsum[r] = 0.f; }

  const int kend = m0 + 16;              // causal: keys < kend
  for (int kb = 0; kb < kend; kb += 32) {
    // scores: per 16-key group load K fragments just-in-time (low VGPR peak)
    v8f s[2];
#pragma unroll
    for (int g = 0; g < 2; ++g) {
      const __bf16* kp = kbase + (size_t)(kb + g * 16 + lm) * KVD;
      v16bf kf0 = *(const v16bf*)(kp + lg * 16);
      v16bf kf1 = *(const v16bf*)(kp + 32 + lg * 16);
      v8f sg = {};
      sg = __builtin_amdgcn_wmma_f32_16x16x32_bf16(false, qa[0].v, false, kf0, (short)0, sg, false, false);
      sg = __builtin_amdgcn_wmma_f32_16x16x32_bf16(false, qa[1].v, false, kf1, (short)0, sg, false, false);
      s[g] = sg;
    }

    const bool needMask = (kb + 31 > m0);

#pragma unroll
    for (int r = 0; r < 8; ++r) {
      int row = m0 + r + 8 * lg;         // C-layout: lanes 0-15 row r, 16-31 row r+8
      float v0 = s[0][r], v1 = s[1][r];
      if (needMask) {
        if (kb + lm      > row) v0 = -__builtin_inff();
        if (kb + 16 + lm > row) v1 = -__builtin_inff();
      }
      // row max over 16 N-lanes (xor of bits 0..3 stays within each half-wave)
      float mx = fmaxf(v0, v1);
#pragma unroll
      for (int off = 1; off < 16; off <<= 1)
        mx = fmaxf(mx, __shfl_xor(mx, off, 32));
      float nm = fmaxf(rmax[r], mx);
      float sc = __expf(rmax[r] - nm);
      rmax[r] = nm;
      float p0 = __expf(v0 - nm);
      float p1 = __expf(v1 - nm);
      float ps = p0 + p1;
#pragma unroll
      for (int off = 1; off < 16; off <<= 1)
        ps += __shfl_xor(ps, off, 32);
      rsum[r] = rsum[r] * sc + ps;
#pragma unroll
      for (int t = 0; t < 4; ++t) acc[t][r] *= sc;
      int lr = r + 8 * lg;
      ldsP[lr * 32 + lm]      = (__bf16)p0;
      ldsP[lr * 32 + 16 + lm] = (__bf16)p1;
    }

    // CDNA5 split counter: make the LDS P tile visible before A-layout reload
    asm volatile("s_wait_dscnt 0" ::: "memory");

    AFrag pa;
    pa.h[0] = *(const v8bf*)(&ldsP[lm * 32 + lg * 16]);
    pa.h[1] = *(const v8bf*)(&ldsP[lm * 32 + lg * 16 + 8]);

#pragma unroll
    for (int t = 0; t < 4; ++t) {
      // Vt B-operand: N = head-dim column, K = 16 contiguous keys per lane
      v16bf vf = *(const v16bf*)(vtbase + (size_t)(t * 16 + lm) * S_LEN + kb + lg * 16);
      acc[t] = __builtin_amdgcn_wmma_f32_16x16x32_bf16(
          false, pa.v, false, vf, (short)0, acc[t], false, false);
    }
  }

#pragma unroll
  for (int t = 0; t < 4; ++t) {
#pragma unroll
    for (int r = 0; r < 8; ++r) {
      int row = b * S_LEN + m0 + r + 8 * lg;
      float val = acc[t][r] / rsum[r];
      o[(size_t)row * EMB + h * HD + t * 16 + lm] = (__bf16)val;
    }
  }
}

// ---------------- host orchestration ----------------
extern "C" void kernel_launch(void* const* d_in, const int* in_sizes, int n_in,
                              void* d_out, int out_size, void* d_ws, size_t ws_size,
                              hipStream_t stream) {
  const float* x  = (const float*)d_in[0];
  const float* Wq = (const float*)d_in[1];
  const float* bq = (const float*)d_in[2];
  const float* Wk = (const float*)d_in[3];
  const float* bk = (const float*)d_in[4];
  const float* Wv = (const float*)d_in[5];
  const float* bv = (const float*)d_in[6];
  const float* Wo = (const float*)d_in[7];
  const float* bo = (const float*)d_in[8];
  float* out = (float*)d_out;

  const int M = 2 * S_LEN;   // 4096 flattened rows

  char* ws = (char*)d_ws;
  size_t off = 0;
  auto alloc = [&](size_t bytes) {
    char* p = ws + off;
    off += (bytes + 255) & ~(size_t)255;
    return p;
  };
  __bf16* xb  = (__bf16*)alloc((size_t)M * EMB * 2);
  __bf16* wqb = (__bf16*)alloc((size_t)EMB * EMB * 2);
  __bf16* wkb = (__bf16*)alloc((size_t)KVD * EMB * 2);
  __bf16* wvb = (__bf16*)alloc((size_t)KVD * EMB * 2);
  __bf16* wob = (__bf16*)alloc((size_t)EMB * EMB * 2);
  __bf16* qb  = (__bf16*)alloc((size_t)M * EMB * 2);
  __bf16* kbf = (__bf16*)alloc((size_t)M * KVD * 2);
  __bf16* vbf = (__bf16*)alloc((size_t)M * KVD * 2);
  __bf16* vtb = (__bf16*)alloc((size_t)M * KVD * 2);
  __bf16* aob = (__bf16*)alloc((size_t)M * EMB * 2);

  auto cast = [&](const float* src, __bf16* dst, int n) {
    cast_f32_bf16<<<(n + 255) / 256, 256, 0, stream>>>(src, dst, n);
  };
  cast(x,  xb,  M * EMB);
  cast(Wq, wqb, EMB * EMB);
  cast(Wk, wkb, KVD * EMB);
  cast(Wv, wvb, KVD * EMB);
  cast(Wo, wob, EMB * EMB);

  // Q projection with 1/sqrt(head_dim) folded in
  gemm_bf16<true><<<dim3(EMB / 64, M / 64), 128, 0, stream>>>(xb, wqb, bq, qb, M, EMB, EMB, 0.125f);
  gemm_bf16<true><<<dim3(KVD / 64, M / 64), 128, 0, stream>>>(xb, wkb, bk, kbf, M, KVD, EMB, 1.0f);
  gemm_bf16<true><<<dim3(KVD / 64, M / 64), 128, 0, stream>>>(xb, wvb, bv, vbf, M, KVD, EMB, 1.0f);

  transpose_v<<<(M * KVD + 255) / 256, 256, 0, stream>>>(vbf, vtb);

  flash_gqa<<<2 * NH * (S_LEN / 16), 32, 0, stream>>>(qb, kbf, vtb, aob);

  // output projection, fp32 result straight to d_out
  gemm_bf16<false><<<dim3(EMB / 64, M / 64), 128, 0, stream>>>(aob, wob, bo, out, M, EMB, EMB, 1.0f);
}
